// CSHead2_89644557402718
// MI455X (gfx1250) — compile-verified, matrix operand order checked by reading, not run on hardware
//
#include <hip/hip_runtime.h>

// ---------------------------------------------------------------------------
// Types
// ---------------------------------------------------------------------------
typedef __bf16 v16bf __attribute__((ext_vector_type(16)));
typedef float  v8f   __attribute__((ext_vector_type(8)));
typedef int    v4i   __attribute__((ext_vector_type(4)));

struct __align__(16) B128 { unsigned int x, y, z, w; };

union FragU { v16bf v; B128 q[2]; };
union OutU  { B128 q; __bf16 h[8]; };

#define HH 56
#define WW 56
#define NPIXIMG (HH * WW)        // 3136 (unpadded, f32 NCHW I/O)
#define PH 58
#define PWID 58
#define NPIXPAD (PH * PWID)      // 3364 (padded NHWC bf16 activations)
#define NB 8

// CDNA5 async global->LDS path (guarded; falls back to sync staging)
#if defined(__gfx1250__) && defined(__has_builtin)
#if __has_builtin(__builtin_amdgcn_global_load_async_to_lds_b128) && \
    __has_builtin(__builtin_amdgcn_s_wait_asynccnt)
#define HAVE_ASYNC 1
#endif
#endif
#ifndef HAVE_ASYNC
#define HAVE_ASYNC 0
#endif

#if HAVE_ASYNC
typedef __attribute__((address_space(1))) v4i* as1_v4i_ptr;
typedef __attribute__((address_space(3))) v4i* as3_v4i_ptr;

__device__ __forceinline__ void async_cp16(const __bf16* g, __bf16* l)
{
    __builtin_amdgcn_global_load_async_to_lds_b128(
        (as1_v4i_ptr)(g),
        (as3_v4i_ptr)(l),
        0, 0);
}
#endif

// ---------------------------------------------------------------------------
// Weight packing: f32 (sel,O,CiReal,taps) -> bf16 WMMA A-fragment layout.
// Fragment f = ((sel*Mtiles + mt)*Ktiles + kt), element (lane, i):
//   M = mt*16 + (lane&15)
//   K = kt*32 + ((i>=8)?16:0) + (lane>>4)*8 + (i&7)     (ISA 16-bit A 16x32)
//   tap = K / CiPad, ci = K % CiPad
// ---------------------------------------------------------------------------
__global__ __launch_bounds__(256)
void pack_w_kernel(const float* __restrict__ w, __bf16* __restrict__ dst,
                   int O, int CiReal, int CiPad, int taps,
                   int Mtiles, int Ktiles, long total)
{
    long idx = (long)blockIdx.x * 256 + threadIdx.x;
    if (idx >= total) return;
    int  i    = idx & 15;
    int  lane = (idx >> 4) & 31;
    long frag = idx >> 9;
    int  kt   = (int)(frag % Ktiles);
    long t2   = frag / Ktiles;
    int  mt   = (int)(t2 % Mtiles);
    long sel  = t2 / Mtiles;

    int M = mt * 16 + (lane & 15);
    int K = kt * 32 + ((i >= 8) ? 16 : 0) + ((lane >> 4) << 3) + (i & 7);
    int tap = K / CiPad;
    int ci  = K % CiPad;

    float v = 0.0f;
    if (M < O && ci < CiReal)
        v = w[(((sel * O + M) * (long)CiReal + ci) * taps) + tap];
    dst[idx] = (__bf16)v;
}

// ---------------------------------------------------------------------------
// Elementwise gates: NCHW f32 features -> padded NHWC bf16 (border untouched)
// ---------------------------------------------------------------------------
__global__ __launch_bounds__(256)
void gate_obj_kernel(const float* __restrict__ feat, const float* __restrict__ emb,
                     __bf16* __restrict__ out)
{
    long i = (long)blockIdx.x * 256 + threadIdx.x;
    if (i >= (long)NB * NPIXIMG * 512) return;
    int  c  = (int)(i & 511);
    long pc = i >> 9;
    int  p  = (int)(pc % NPIXIMG);
    int  b  = (int)(pc / NPIXIMG);
    int  h  = p / WW, w = p % WW;
    float v = feat[((long)b * 512 + c) * NPIXIMG + p] * emb[16 * 512 + c];
    out[((long)b * NPIXPAD + (h + 1) * PWID + (w + 1)) * 512 + c] = (__bf16)v;
}

__global__ __launch_bounds__(256)
void gate_part_kernel(const float* __restrict__ feat, const float* __restrict__ emb,
                      const int* __restrict__ instr, const __bf16* __restrict__ bu,
                      __bf16* __restrict__ out)
{
    long i = (long)blockIdx.x * 256 + threadIdx.x;
    if (i >= (long)NB * NPIXIMG * 512) return;
    int  c  = (int)(i & 511);
    long pc = i >> 9;
    int  p  = (int)(pc % NPIXIMG);
    int  b  = (int)(pc / NPIXIMG);
    int  h  = p / WW, w = p % WW;
    long pidx = ((long)b * NPIXPAD + (h + 1) * PWID + (w + 1)) * 512 + c;
    float v = feat[((long)b * 512 + c) * NPIXIMG + p] * emb[(long)instr[b] * 512 + c]
            + (float)bu[pidx];
    out[pidx] = (__bf16)v;
}

// ---------------------------------------------------------------------------
// Implicit-GEMM conv via WMMA bf16 (TAPS = 9 -> 3x3 SAME, TAPS = 1 -> 1x1).
// Input: zero-padded NHWC bf16 (58x58, stride CI) -> staging is branch-free.
// Workgroup: 4 waves = 8x8 pixel block x 64 co; each wave 16 co x 4 N-tiles.
// Double-buffered LDS; async global->LDS staging when available.
// ---------------------------------------------------------------------------
template <int TAPS, int CI>
__global__ __launch_bounds__(128)
void conv_wmma_kernel(const __bf16* __restrict__ in,
                      const __bf16* __restrict__ wpk, long wSelStride,
                      const float*  __restrict__ bias, int biasLen, int biasSelStride,
                      const int*    __restrict__ instr,
                      int Co, int Ktiles,
                      __bf16* __restrict__ outb, int CoutStride, int coutOff,
                      float*  __restrict__ outf, int CoF32,
                      int relu)
{
    constexpr int HALO  = (TAPS == 9) ? 1 : 0;
    constexpr int TW    = 8 + 2 * HALO;        // 10 or 8
    constexpr int NPIX  = TW * TW;             // 100 or 64
    constexpr int CCH   = (CI < 64) ? CI : 64; // LDS channel chunk
    constexpr int NCH   = CI / CCH;            // number of chunks
    constexpr int TASKS = NPIX * (CCH / 8);    // 16B copies per chunk
    constexpr int KITER = (TASKS + 127) / 128; // wave-uniform issue count

    __shared__ __bf16 smem[2][NPIX * CCH];

    const int b    = blockIdx.z;
    const int bh   = (blockIdx.x / 7) * 8;
    const int bw   = (blockIdx.x % 7) * 8;
    const int tid  = threadIdx.x;
    const int lane = tid & 31;
    const int wave = tid >> 5;
    const int mt   = blockIdx.y * 4 + wave;
    const bool active = (mt * 16) < Co;

    const __bf16* wsel = wpk;
    const float*  bsel = bias;
    if (instr) {
        int s = instr[b];
        wsel += (long)s * wSelStride;
        if (bsel) bsel += (long)s * biasSelStride;
    }

    v8f acc[4];
#pragma unroll
    for (int nt = 0; nt < 4; ++nt)
#pragma unroll
        for (int r = 0; r < 8; ++r) acc[nt][r] = 0.0f;

    const long pixBase = (long)b * NPIXPAD;

    // branch-free copy-task address (always in padded bounds)
    auto task_src = [&](int t, int c0) -> const __bf16* {
        int sp = t / (CCH / 8);
        int cc = (t - sp * (CCH / 8)) * 8;
        int gh = bh + sp / TW - HALO + 1;   // padded row in [0,57]
        int gw = bw + sp % TW - HALO + 1;
        return in + ((pixBase + gh * PWID + gw) * (long)CI + c0 + cc);
    };
    auto task_dst = [&](int t, int buf) -> __bf16* {
        int sp = t / (CCH / 8);
        int cc = (t - sp * (CCH / 8)) * 8;
        return &smem[buf][sp * CCH + cc];
    };

#if HAVE_ASYNC
    auto issue = [&](int c0, int buf) {
#pragma unroll
        for (int k = 0; k < KITER; ++k) {
            int t = tid + k * 128;
            if (t >= TASKS) t = TASKS - 1;   // duplicate tail task (benign)
            async_cp16(task_src(t, c0), task_dst(t, buf));
        }
    };
    issue(0, 0);
#endif

    for (int ci = 0; ci < NCH; ++ci) {
        const int c0  = ci * CCH;
        const int cur = ci & 1;
#if HAVE_ASYNC
        if (ci > 0) __syncthreads();             // buf (ci+1)&1 free to overwrite
        if (ci + 1 < NCH) {
            issue((ci + 1) * CCH, (ci + 1) & 1);
            __builtin_amdgcn_s_wait_asynccnt(KITER); // chunk ci complete (in order)
        } else {
            __builtin_amdgcn_s_wait_asynccnt(0);
        }
        __syncthreads();                         // chunk ci visible to all waves
#else
        __syncthreads();
#pragma unroll
        for (int k = 0; k < KITER; ++k) {
            int t = tid + k * 128;
            if (t >= TASKS) t = TASKS - 1;
            *reinterpret_cast<B128*>(task_dst(t, cur)) =
                *reinterpret_cast<const B128*>(task_src(t, c0));
        }
        __syncthreads();
#endif
        if (!active) continue;

#pragma unroll
        for (int tap = 0; tap < TAPS; ++tap) {
            const int dh = (TAPS == 9) ? (tap / 3) : 0;   // halo-relative
            const int dw = (TAPS == 9) ? (tap % 3) : 0;
#pragma unroll
            for (int kk = 0; kk < CCH; kk += 32) {
                const int ktg = (tap * CI + c0 + kk) >> 5;
                // A fragment: pre-packed, 2 x b128 per lane, reused over 4 N-tiles
                FragU fa;
                const __bf16* ap = wsel + ((long)(mt * Ktiles + ktg) << 9) + lane * 16;
                fa.q[0] = *reinterpret_cast<const B128*>(ap);
                fa.q[1] = *reinterpret_cast<const B128*>(ap + 8);
#pragma unroll
                for (int nt = 0; nt < 4; ++nt) {
                    const int px = nt * 16 + (lane & 15);
                    const int sp = ((px >> 3) + dh) * TW + (px & 7) + dw;
                    FragU fb;
                    const __bf16* bp = &smem[cur][sp * CCH + kk + ((lane >> 4) << 3)];
                    fb.q[0] = *reinterpret_cast<const B128*>(bp);
                    fb.q[1] = *reinterpret_cast<const B128*>(bp + 16);
                    acc[nt] = __builtin_amdgcn_wmma_f32_16x16x32_bf16(
                        false, fa.v, false, fb.v, (short)0, acc[nt], false, false);
                }
            }
        }
    }

    if (!active) return;

    // ---- epilogue: bias + relu, bf16 padded-NHWC store (b128) / f32 NCHW ----
    const int half   = lane >> 4;
    const int cobase = mt * 16 + half * 8;
    float bvals[8];
#pragma unroll
    for (int r = 0; r < 8; ++r) {
        int co = cobase + r;
        bvals[r] = (bias && co < biasLen) ? bsel[co] : 0.0f;
    }
#pragma unroll
    for (int nt = 0; nt < 4; ++nt) {
        const int px = nt * 16 + (lane & 15);
        const int oh = bh + (px >> 3), ow = bw + (px & 7);
        if (outb) {
            OutU o;
#pragma unroll
            for (int r = 0; r < 8; ++r) {
                float v = acc[nt][r] + bvals[r];
                if (relu) v = v > 0.0f ? v : 0.0f;
                o.h[r] = (__bf16)v;
            }
            long opixp = (long)(oh + 1) * PWID + (ow + 1);
            *reinterpret_cast<B128*>(
                outb + ((pixBase + opixp) * (long)CoutStride + coutOff + cobase)) = o.q;
        }
        if (outf) {
            const int opix = oh * WW + ow;
#pragma unroll
            for (int r = 0; r < 8; ++r) {
                int co = cobase + r;
                if (co < CoF32) {
                    float v = acc[nt][r] + bvals[r];
                    if (relu) v = v > 0.0f ? v : 0.0f;
                    outf[((long)b * CoF32 + co) * NPIXIMG + opix] = v;
                }
            }
        }
    }
}

// ---------------------------------------------------------------------------
// Host-side orchestration
// ---------------------------------------------------------------------------
extern "C" void kernel_launch(void* const* d_in, const int* in_sizes, int n_in,
                              void* d_out, int out_size, void* d_ws, size_t ws_size,
                              hipStream_t stream)
{
    (void)in_sizes; (void)n_in; (void)out_size; (void)ws_size;

    const float* features = (const float*)d_in[0];
    const int*   instr    = (const int*)  d_in[1];
    const float* emb      = (const float*)d_in[2];
    const float* td_w1    = (const float*)d_in[3];
    const float* td_b1    = (const float*)d_in[4];
    const float* td_w2    = (const float*)d_in[5];
    const float* td_b2    = (const float*)d_in[6];
    const float* w_obj    = (const float*)d_in[7];
    const float* b_obj    = (const float*)d_in[8];
    const float* w_bus    = (const float*)d_in[9];
    const float* w_lat1   = (const float*)d_in[10];
    const float* b_lat1   = (const float*)d_in[11];
    const float* w_lat2   = (const float*)d_in[12];
    const float* b_lat2   = (const float*)d_in[13];
    const float* w_bu1    = (const float*)d_in[14];
    const float* b_bu1    = (const float*)d_in[15];
    const float* w_bu2    = (const float*)d_in[16];
    const float* b_bu2    = (const float*)d_in[17];
    const float* w_part   = (const float*)d_in[18];
    const float* b_part   = (const float*)d_in[19];

    float* out_obj  = (float*)d_out;                           // [8,21,56,56]
    float* out_part = (float*)d_out + (long)NB * 21 * NPIXIMG; // [8,8,56,56]

    // --- scratch partition ---
    char*  ws  = (char*)d_ws;
    size_t off = 0;
    auto alloc = [&](size_t bytes) -> char* {
        char* p = ws + off;
        off = (off + bytes + 255) & ~(size_t)255;
        return p;
    };
    const size_t ACT = (size_t)NB * NPIXPAD * 512 * sizeof(__bf16);
    __bf16* bufA = (__bf16*)alloc(ACT);   // X0 -> later BU
    __bf16* bufB = (__bf16*)alloc(ACT);   // T1
    __bf16* bufC = (__bf16*)alloc(ACT);   // T2 -> later Y0
    __bf16* bufD = (__bf16*)alloc(ACT);   // XCAT1 -> later T1'
    __bf16* bufE = (__bf16*)alloc(ACT);   // XCAT2 -> later T2'
    __bf16* bufOBJ = (__bf16*)alloc((size_t)NB * NPIXPAD * 32 * sizeof(__bf16));
    const size_t actBytes = off;          // contiguous activation region

    auto packed = [&](int Mtiles, int Ktiles, int nSel) -> __bf16* {
        return (__bf16*)alloc((size_t)nSel * Mtiles * Ktiles * 512 * sizeof(__bf16));
    };
    __bf16* PW1  = packed(32, 144, 1);   // td_w1   512<-512, 3x3
    __bf16* PW2  = packed(32, 144, 1);   // td_w2
    __bf16* PWOH = packed(2, 16, 1);     // obj head 21(pad32)<-512, 1x1
    __bf16* PWBS = packed(16, 9, 1);     // bu_start 256<-21(pad32), 3x3
    __bf16* PWL1 = packed(16, 144, 1);   // lat1 256<-512
    __bf16* PWL2 = packed(16, 144, 1);   // lat2
    __bf16* PWB1 = packed(16, 144, 1);   // bu1 256<-512
    __bf16* PWB2 = packed(32, 144, 1);   // bu2 512<-512
    __bf16* PWPT = packed(1, 16, 16);    // part heads: 16 sels x (8(pad16)<-512)

    // zero activation region once per launch (borders must be zero for SAME pad)
    (void)hipMemsetAsync(bufA, 0, actBytes, stream);

    // --- weight packing ---
    auto pack = [&](const float* w, __bf16* dst, int O, int CiReal, int CiPad,
                    int taps, int Mtiles, int Ktiles, int nSel) {
        long total = (long)nSel * Mtiles * Ktiles * 512;
        int  grid  = (int)((total + 255) / 256);
        pack_w_kernel<<<grid, 256, 0, stream>>>(w, dst, O, CiReal, CiPad, taps,
                                                Mtiles, Ktiles, total);
    };
    pack(td_w1,  PW1,  512, 512, 512, 9, 32, 144, 1);
    pack(td_w2,  PW2,  512, 512, 512, 9, 32, 144, 1);
    pack(w_obj,  PWOH,  21, 512, 512, 1,  2,  16, 1);
    pack(w_bus,  PWBS, 256,  21,  32, 9, 16,   9, 1);
    pack(w_lat1, PWL1, 256, 512, 512, 9, 16, 144, 1);
    pack(w_lat2, PWL2, 256, 512, 512, 9, 16, 144, 1);
    pack(w_bu1,  PWB1, 256, 512, 512, 9, 16, 144, 1);
    pack(w_bu2,  PWB2, 512, 512, 512, 9, 32, 144, 1);
    pack(w_part, PWPT,   8, 512, 512, 1,  1,  16, 16);

    const long nElem = (long)NB * NPIXIMG * 512;
    const int  gGate = (int)((nElem + 255) / 256);

    // ---- object pathway ----
    gate_obj_kernel<<<gGate, 256, 0, stream>>>(features, emb, bufA);  // X0

    dim3 blk(128);
    auto grid_of = [](int Co) { return dim3(49, (Co + 63) / 64, NB); };

    // t1 = relu(conv3x3(X0, td_w1)+b1)
    conv_wmma_kernel<9, 512><<<grid_of(512), blk, 0, stream>>>(
        bufA, PW1, 0, td_b1, 512, 0, nullptr, 512, 144,
        bufB, 512, 0, nullptr, 0, 1);
    // t2 = relu(conv3x3(t1, td_w2)+b2)
    conv_wmma_kernel<9, 512><<<grid_of(512), blk, 0, stream>>>(
        bufB, PW2, 0, td_b2, 512, 0, nullptr, 512, 144,
        bufC, 512, 0, nullptr, 0, 1);
    // obj_pred = 1x1 head -> d_out (f32 NCHW) + bf16 padded NHWC (32ch)
    conv_wmma_kernel<1, 512><<<grid_of(32), blk, 0, stream>>>(
        bufC, PWOH, 0, b_obj, 21, 0, nullptr, 32, 16,
        bufOBJ, 32, 0, out_obj, 21, 0);
    // bu_start: conv3x3(obj_pred) -> XCAT1[0:256)
    conv_wmma_kernel<9, 32><<<grid_of(256), blk, 0, stream>>>(
        bufOBJ, PWBS, 0, nullptr, 0, 0, nullptr, 256, 9,
        bufD, 512, 0, nullptr, 0, 0);
    // lat1: relu(conv3x3(t1)) -> XCAT1[256:512)
    conv_wmma_kernel<9, 512><<<grid_of(256), blk, 0, stream>>>(
        bufB, PWL1, 0, b_lat1, 256, 0, nullptr, 256, 144,
        bufD, 512, 256, nullptr, 0, 1);
    // bu1: relu(conv3x3(XCAT1)) -> XCAT2[0:256)
    conv_wmma_kernel<9, 512><<<grid_of(256), blk, 0, stream>>>(
        bufD, PWB1, 0, b_bu1, 256, 0, nullptr, 256, 144,
        bufE, 512, 0, nullptr, 0, 1);
    // lat2: relu(conv3x3(t2)) -> XCAT2[256:512)
    conv_wmma_kernel<9, 512><<<grid_of(256), blk, 0, stream>>>(
        bufC, PWL2, 0, b_lat2, 256, 0, nullptr, 256, 144,
        bufE, 512, 256, nullptr, 0, 1);
    // bu = relu(conv3x3(XCAT2, w_bu2)+b) -> bufA (X0 dead)
    conv_wmma_kernel<9, 512><<<grid_of(512), blk, 0, stream>>>(
        bufE, PWB2, 0, b_bu2, 512, 0, nullptr, 512, 144,
        bufA, 512, 0, nullptr, 0, 1);

    // ---- part pathway ----
    // y0 = features*emb[instr] + bu  -> bufC (T2 dead)
    gate_part_kernel<<<gGate, 256, 0, stream>>>(features, emb, instr, bufA, bufC);
    // t1' -> bufD (XCAT1 dead)
    conv_wmma_kernel<9, 512><<<grid_of(512), blk, 0, stream>>>(
        bufC, PW1, 0, td_b1, 512, 0, nullptr, 512, 144,
        bufD, 512, 0, nullptr, 0, 1);
    // t2' -> bufE (XCAT2 dead)
    conv_wmma_kernel<9, 512><<<grid_of(512), blk, 0, stream>>>(
        bufD, PW2, 0, td_b2, 512, 0, nullptr, 512, 144,
        bufE, 512, 0, nullptr, 0, 1);
    // part_pred: per-sample 1x1 head (instruction-selected weights/bias) -> d_out
    conv_wmma_kernel<1, 512><<<grid_of(16), blk, 0, stream>>>(
        bufE, PWPT, /*wSelStride=*/16 * 512, b_part, 8, /*biasSelStride=*/8,
        instr, 16, 16,
        nullptr, 16, 0, out_part, 8, 0);
}
